// CausalSelfAttention_7310034338544
// MI455X (gfx1250) — compile-verified, matrix operand order checked by reading, not run on hardware
//
#include <hip/hip_runtime.h>
#include <hip/hip_bf16.h>

// ---------------------------------------------------------------------------
// CDNA5 (gfx1250) causal self-attention forward.
// All matmuls run on v_wmma_f32_16x16x32_bf16 (bf16 operands, f32 accum).
// GEMMs are register-staged double-buffered; attention K-tile uses
// GLOBAL_LOAD_ASYNC_TO_LDS_B128 (ASYNCcnt-tracked DMA, no VGPR bounce).
// ---------------------------------------------------------------------------

typedef __attribute__((ext_vector_type(16))) __bf16 v16bf;
typedef __attribute__((ext_vector_type(8)))  float  v8f;
typedef __attribute__((ext_vector_type(4)))  int    v4i;

union BFrag {
    v16bf v;
    uint4 u4[2];
};

__device__ __forceinline__ v8f wmma_bf16(v16bf a, v16bf b, v8f c) {
    // 8 args: (neg_a, A, neg_b, B, c_mod, C, reuse_a, reuse_b)
    return __builtin_amdgcn_wmma_f32_16x16x32_bf16(false, a, false, b,
                                                   (short)0, c, false, false);
}

#if __has_builtin(__builtin_amdgcn_global_load_async_to_lds_b128)
#define HAVE_ASYNC_LDS 1
typedef __attribute__((address_space(1))) v4i gv4i;   // global int4
typedef __attribute__((address_space(3))) v4i lv4i;   // LDS int4
#endif

#define D_MODEL   1024
#define N_HEADS   16
#define HEAD_DIM  64
#define BATCH     4
#define SEQ       2048
#define M_TOTAL   (BATCH * SEQ)          // 8192
#define QKV_N     (3 * D_MODEL)          // 3072
#define SEL_STRIDE ((size_t)BATCH * N_HEADS * SEQ * HEAD_DIM)  // 8388608 elems

// ===========================================================================
// Kernel 1: QKV projection.  C[8192,3072] = X[8192,1024] * Wqkv[1024,3072]+b.
// 128x128 block tile, BK=32, 8 waves (4 row-groups x 2 col-groups),
// register-staged LDS double buffering: one barrier per K-step, global loads
// for tile i+1 overlap the 8 WMMAs of tile i.
// Epilogue scatters bf16 into head-major [sel][b][h][t][d] workspace.
// ===========================================================================
__global__ __launch_bounds__(256) void gemm_qkv_kernel(
    const float* __restrict__ X, const float* __restrict__ W,
    const float* __restrict__ bias, __bf16* __restrict__ qkvws)
{
    __shared__ __bf16 As[2][128][32];   // [buf][m][k]
    __shared__ __bf16 Bs[2][128][32];   // [buf][n][k] (W tile transposed)

    const int n0 = blockIdx.x * 128;
    const int m0 = blockIdx.y * 128;
    const int tid  = threadIdx.x;
    const int lane = tid & 31, wid = tid >> 5;
    const int hi = lane >> 4, ln = lane & 15;
    const int wr = wid & 3, wc = wid >> 2;

    v8f acc[2][4];
#pragma unroll
    for (int r = 0; r < 2; ++r)
#pragma unroll
        for (int c = 0; c < 4; ++c)
#pragma unroll
            for (int i = 0; i < 8; ++i) acc[r][c][i] = 0.0f;

    float4 aReg[4], bReg[4];
    // Preload tile k0 = 0 into registers.
#pragma unroll
    for (int i = 0; i < 4; ++i) {
        int idx = tid + i * 256;
        int row = idx >> 3, kq = (idx & 7) * 4;
        aReg[i] = *(const float4*)(X + (size_t)(m0 + row) * D_MODEL + kq);
        int kr = idx >> 5, nq = (idx & 31) * 4;
        bReg[i] = *(const float4*)(W + (size_t)kr * QKV_N + n0 + nq);
    }

    int p = 0;
    for (int k0 = 0; k0 < D_MODEL; k0 += 32) {
        // Commit staged registers (tile k0) to LDS buffer p.
#pragma unroll
        for (int i = 0; i < 4; ++i) {
            int idx = tid + i * 256;
            int row = idx >> 3, kq = (idx & 7) * 4;
            As[p][row][kq + 0] = (__bf16)aReg[i].x;
            As[p][row][kq + 1] = (__bf16)aReg[i].y;
            As[p][row][kq + 2] = (__bf16)aReg[i].z;
            As[p][row][kq + 3] = (__bf16)aReg[i].w;
            int kr = idx >> 5, nq = (idx & 31) * 4;
            Bs[p][nq + 0][kr] = (__bf16)bReg[i].x;
            Bs[p][nq + 1][kr] = (__bf16)bReg[i].y;
            Bs[p][nq + 2][kr] = (__bf16)bReg[i].z;
            Bs[p][nq + 3][kr] = (__bf16)bReg[i].w;
        }
        __syncthreads();

        // Issue global loads for tile k0+32 (overlap with WMMAs below).
        if (k0 + 32 < D_MODEL) {
#pragma unroll
            for (int i = 0; i < 4; ++i) {
                int idx = tid + i * 256;
                int row = idx >> 3, kq = (idx & 7) * 4;
                aReg[i] = *(const float4*)(X + (size_t)(m0 + row) * D_MODEL +
                                           k0 + 32 + kq);
                int kr = idx >> 5, nq = (idx & 31) * 4;
                bReg[i] = *(const float4*)(W + (size_t)(k0 + 32 + kr) * QKV_N +
                                           n0 + nq);
            }
            // Warm L2 for the tile after that (global_prefetch_b8).
            if (k0 + 64 < D_MODEL) {
                int row = tid >> 3, kq = (tid & 7) * 4;
                __builtin_prefetch(X + (size_t)(m0 + row) * D_MODEL + k0 + 64 + kq, 0, 1);
                int kr = tid >> 5, nq = (tid & 31) * 4;
                __builtin_prefetch(W + (size_t)(k0 + 64 + kr) * QKV_N + n0 + nq, 0, 1);
            }
        }

        // Compute on LDS buffer p.
        BFrag af[2], bf[4];
#pragma unroll
        for (int r = 0; r < 2; ++r) {
            const __bf16* ap = &As[p][wr * 32 + r * 16 + ln][0];
            af[r].u4[0] = *(const uint4*)(ap + hi * 8);
            af[r].u4[1] = *(const uint4*)(ap + 16 + hi * 8);
        }
#pragma unroll
        for (int c = 0; c < 4; ++c) {
            const __bf16* bp = &Bs[p][wc * 64 + c * 16 + ln][0];
            bf[c].u4[0] = *(const uint4*)(bp + hi * 8);
            bf[c].u4[1] = *(const uint4*)(bp + 16 + hi * 8);
        }
#pragma unroll
        for (int r = 0; r < 2; ++r)
#pragma unroll
            for (int c = 0; c < 4; ++c)
                acc[r][c] = wmma_bf16(af[r].v, bf[c].v, acc[r][c]);

        p ^= 1;
    }

    // Epilogue: bias + scatter to [sel][b][h][t][d] bf16
#pragma unroll
    for (int r = 0; r < 2; ++r)
#pragma unroll
        for (int c = 0; c < 4; ++c)
#pragma unroll
            for (int i = 0; i < 8; ++i) {
                int m = m0 + wr * 32 + r * 16 + i + hi * 8;
                int n = n0 + wc * 64 + c * 16 + ln;
                float v = acc[r][c][i] + bias[n];
                int sel = n >> 10, cc = n & 1023;
                int h = cc >> 6, d = cc & 63;
                int b = m >> 11, t = m & 2047;
                qkvws[(((size_t)(sel * BATCH + b) * N_HEADS + h) * SEQ + t) * HEAD_DIM + d]
                    = (__bf16)v;
            }
}

// ===========================================================================
// Kernel 2: RoPE (in place, bf16). One thread per (b,h,t,d<32) pair.
// ===========================================================================
__global__ __launch_bounds__(256) void rope_kernel(__bf16* __restrict__ buf)
{
    int idx = blockIdx.x * blockDim.x + threadIdx.x;   // B*H*T*32 threads
    int d  = idx & 31;
    int t  = (idx >> 5) & 2047;
    int bh = idx >> 16;
    size_t base = ((size_t)bh * SEQ + t) * HEAD_DIM;
    // inv_freq = 10000^(-2d/64) = exp(-(d/32)*ln(10000))
    float ang = (float)t * __expf(-(float)d * 0.03125f * 9.210340371976184f);
    float s, c;
    __sincosf(ang, &s, &c);
    float x1 = (float)buf[base + d];
    float x2 = (float)buf[base + 32 + d];
    buf[base + d]      = (__bf16)(x1 * c - x2 * s);
    buf[base + 32 + d] = (__bf16)(x2 * c + x1 * s);
}

// ===========================================================================
// Kernel 3: flash attention. Grid (B*H, T/128). 8 waves x 16 query rows.
// S = Q*K^T (4 WMMAs), online softmax, P*V (4 WMMAs) per 32-key tile.
// K tile is staged with GLOBAL_LOAD_ASYNC_TO_LDS_B128 when available.
// ===========================================================================
__global__ __launch_bounds__(256) void attn_kernel(
    const __bf16* __restrict__ qkvws, __bf16* __restrict__ attnout)
{
    __shared__ __bf16 Kl[32][64];      // [key][d]
    __shared__ __bf16 Vl[64][32];      // [d][key]  (transposed for B-frag)
    __shared__ __bf16 Pl[8][16][32];   // per-wave P transpose buffer

    const int bh = blockIdx.x;
    const int b = bh >> 4, h = bh & 15;
    const int q0 = blockIdx.y * 128;
    const int tid = threadIdx.x, lane = tid & 31, wid = tid >> 5;
    const int hi = lane >> 4, ln = lane & 15;
    const int qw = q0 + wid * 16;

    const __bf16* Qp = qkvws + (size_t)bh * SEQ * HEAD_DIM;
    const __bf16* Kp = qkvws + SEL_STRIDE + (size_t)bh * SEQ * HEAD_DIM;
    const __bf16* Vp = qkvws + 2 * SEL_STRIDE + (size_t)bh * SEQ * HEAD_DIM;

    // Q fragments (A layout, d-chunks 0..31 and 32..63) straight from global.
    BFrag qf[2];
    {
        const __bf16* qrow = Qp + (size_t)(qw + ln) * HEAD_DIM;
        qf[0].u4[0] = *(const uint4*)(qrow + hi * 8);
        qf[0].u4[1] = *(const uint4*)(qrow + 16 + hi * 8);
        qf[1].u4[0] = *(const uint4*)(qrow + 32 + hi * 8);
        qf[1].u4[1] = *(const uint4*)(qrow + 48 + hi * 8);
    }

    v8f acc[4];
    float mrun[8], lrun[8];
#pragma unroll
    for (int c = 0; c < 4; ++c)
#pragma unroll
        for (int i = 0; i < 8; ++i) acc[c][i] = 0.0f;
#pragma unroll
    for (int i = 0; i < 8; ++i) { mrun[i] = -3.0e38f; lrun[i] = 0.0f; }

    for (int kt0 = 0; kt0 < q0 + 128; kt0 += 32) {
        // Cooperative stage of K (row-major) and V (transposed) tiles.
        {
            int key = tid >> 3, dq = (tid & 7) * 8;
#ifdef HAVE_ASYNC_LDS
            // ASYNCcnt-tracked DMA: global -> LDS, no VGPR bounce.
            __builtin_amdgcn_global_load_async_to_lds_b128(
                (gv4i*)(Kp + (size_t)(kt0 + key) * HEAD_DIM + dq),
                (lv4i*)&Kl[key][dq], 0, 0);
#else
            *(uint4*)&Kl[key][dq] =
                *(const uint4*)(Kp + (size_t)(kt0 + key) * HEAD_DIM + dq);
#endif
            union { uint4 q; __bf16 e[8]; } cv;
            cv.q = *(const uint4*)(Vp + (size_t)(kt0 + key) * HEAD_DIM + dq);
#pragma unroll
            for (int j = 0; j < 8; ++j) Vl[dq + j][key] = cv.e[j];
#ifdef HAVE_ASYNC_LDS
#if __has_builtin(__builtin_amdgcn_s_wait_asynccnt)
            __builtin_amdgcn_s_wait_asynccnt(0);
#else
            asm volatile("s_wait_asynccnt 0" ::: "memory");
#endif
#endif
        }
        __syncthreads();

        if (kt0 <= qw + 15) {          // wave-uniform: EXEC stays all-1s
            v8f s[2];
#pragma unroll
            for (int g = 0; g < 2; ++g)
#pragma unroll
                for (int i = 0; i < 8; ++i) s[g][i] = 0.0f;

#pragma unroll
            for (int g = 0; g < 2; ++g)
#pragma unroll
                for (int kk = 0; kk < 2; ++kk) {
                    BFrag kf;
                    const __bf16* krow = &Kl[g * 16 + ln][kk * 32 + hi * 8];
                    kf.u4[0] = *(const uint4*)(krow);
                    kf.u4[1] = *(const uint4*)(krow + 16);
                    s[g] = wmma_bf16(qf[kk].v, kf.v, s[g]);
                }

            // Online softmax: rows live at VGPR i -> row = i + 8*hi;
            // cols striped over 16 lanes -> reduce with xor 1,2,4,8.
#pragma unroll
            for (int i = 0; i < 8; ++i) {
                int qi = qw + i + hi * 8;
                float a0 = s[0][i] * 0.125f;         // 1/sqrt(64)
                float a1 = s[1][i] * 0.125f;
                if (kt0 + ln > qi)      a0 = -3.0e38f;
                if (kt0 + 16 + ln > qi) a1 = -3.0e38f;
                float tm = fmaxf(a0, a1);
                tm = fmaxf(tm, __shfl_xor(tm, 1, 32));
                tm = fmaxf(tm, __shfl_xor(tm, 2, 32));
                tm = fmaxf(tm, __shfl_xor(tm, 4, 32));
                tm = fmaxf(tm, __shfl_xor(tm, 8, 32));
                float newm = fmaxf(mrun[i], tm);
                float sc = __expf(mrun[i] - newm);
                mrun[i] = newm;
                float p0 = __expf(a0 - newm);
                float p1 = __expf(a1 - newm);
                float ts = p0 + p1;
                ts += __shfl_xor(ts, 1, 32);
                ts += __shfl_xor(ts, 2, 32);
                ts += __shfl_xor(ts, 4, 32);
                ts += __shfl_xor(ts, 8, 32);
                lrun[i] = lrun[i] * sc + ts;
#pragma unroll
                for (int c = 0; c < 4; ++c) acc[c][i] *= sc;
                Pl[wid][i + hi * 8][ln]      = (__bf16)p0;
                Pl[wid][i + hi * 8][16 + ln] = (__bf16)p1;
            }
            // same-wave DS RAW: make P stores visible before frag reads
            asm volatile("s_wait_dscnt 0" ::: "memory");

            BFrag pf;
            const __bf16* prow = &Pl[wid][ln][0];
            pf.u4[0] = *(const uint4*)(prow + hi * 8);
            pf.u4[1] = *(const uint4*)(prow + 16 + hi * 8);
#pragma unroll
            for (int c = 0; c < 4; ++c) {
                BFrag vf;
                const __bf16* vrow = &Vl[c * 16 + ln][0];
                vf.u4[0] = *(const uint4*)(vrow + hi * 8);
                vf.u4[1] = *(const uint4*)(vrow + 16 + hi * 8);
                acc[c] = wmma_bf16(pf.v, vf.v, acc[c]);
            }
        }
        __syncthreads();
    }

    // Normalize and scatter back to [B,T,C] (bf16) for the out-proj GEMM.
#pragma unroll
    for (int c = 0; c < 4; ++c)
#pragma unroll
        for (int i = 0; i < 8; ++i) {
            int t = qw + i + hi * 8;
            float v = acc[c][i] / lrun[i];
            attnout[((size_t)(b * SEQ + t)) * D_MODEL + h * HEAD_DIM + c * 16 + ln]
                = (__bf16)v;
        }
}

// ===========================================================================
// Kernel 4: output projection. out[8192,1024] = A_bf16 * Wout + bout (f32 out)
// Same double-buffered pipeline as kernel 1; A is already bf16.
// ===========================================================================
__global__ __launch_bounds__(256) void gemm_out_kernel(
    const __bf16* __restrict__ A, const float* __restrict__ W,
    const float* __restrict__ bias, float* __restrict__ out)
{
    __shared__ __bf16 As[2][128][32];
    __shared__ __bf16 Bs[2][128][32];

    const int n0 = blockIdx.x * 128;
    const int m0 = blockIdx.y * 128;
    const int tid  = threadIdx.x;
    const int lane = tid & 31, wid = tid >> 5;
    const int hi = lane >> 4, ln = lane & 15;
    const int wr = wid & 3, wc = wid >> 2;

    v8f acc[2][4];
#pragma unroll
    for (int r = 0; r < 2; ++r)
#pragma unroll
        for (int c = 0; c < 4; ++c)
#pragma unroll
            for (int i = 0; i < 8; ++i) acc[r][c][i] = 0.0f;

    uint2  aReg[4];
    float4 bReg[4];
#pragma unroll
    for (int i = 0; i < 4; ++i) {
        int idx = tid + i * 256;
        int row = idx >> 3, kq = (idx & 7) * 4;
        aReg[i] = *(const uint2*)(A + (size_t)(m0 + row) * D_MODEL + kq);
        int kr = idx >> 5, nq = (idx & 31) * 4;
        bReg[i] = *(const float4*)(W + (size_t)kr * D_MODEL + n0 + nq);
    }

    int p = 0;
    for (int k0 = 0; k0 < D_MODEL; k0 += 32) {
#pragma unroll
        for (int i = 0; i < 4; ++i) {
            int idx = tid + i * 256;
            int row = idx >> 3, kq = (idx & 7) * 4;
            *(uint2*)&As[p][row][kq] = aReg[i];
            int kr = idx >> 5, nq = (idx & 31) * 4;
            Bs[p][nq + 0][kr] = (__bf16)bReg[i].x;
            Bs[p][nq + 1][kr] = (__bf16)bReg[i].y;
            Bs[p][nq + 2][kr] = (__bf16)bReg[i].z;
            Bs[p][nq + 3][kr] = (__bf16)bReg[i].w;
        }
        __syncthreads();

        if (k0 + 32 < D_MODEL) {
#pragma unroll
            for (int i = 0; i < 4; ++i) {
                int idx = tid + i * 256;
                int row = idx >> 3, kq = (idx & 7) * 4;
                aReg[i] = *(const uint2*)(A + (size_t)(m0 + row) * D_MODEL +
                                          k0 + 32 + kq);
                int kr = idx >> 5, nq = (idx & 31) * 4;
                bReg[i] = *(const float4*)(W + (size_t)(k0 + 32 + kr) * D_MODEL +
                                           n0 + nq);
            }
            if (k0 + 64 < D_MODEL) {
                int row = tid >> 3, kq = (tid & 7) * 4;
                __builtin_prefetch(A + (size_t)(m0 + row) * D_MODEL + k0 + 64 + kq, 0, 1);
                int kr = tid >> 5, nq = (tid & 31) * 4;
                __builtin_prefetch(W + (size_t)(k0 + 64 + kr) * D_MODEL + n0 + nq, 0, 1);
            }
        }

        BFrag af[2], bf[4];
#pragma unroll
        for (int r = 0; r < 2; ++r) {
            const __bf16* ap = &As[p][wr * 32 + r * 16 + ln][0];
            af[r].u4[0] = *(const uint4*)(ap + hi * 8);
            af[r].u4[1] = *(const uint4*)(ap + 16 + hi * 8);
        }
#pragma unroll
        for (int c = 0; c < 4; ++c) {
            const __bf16* bp = &Bs[p][wc * 64 + c * 16 + ln][0];
            bf[c].u4[0] = *(const uint4*)(bp + hi * 8);
            bf[c].u4[1] = *(const uint4*)(bp + 16 + hi * 8);
        }
#pragma unroll
        for (int r = 0; r < 2; ++r)
#pragma unroll
            for (int c = 0; c < 4; ++c)
                acc[r][c] = wmma_bf16(af[r].v, bf[c].v, acc[r][c]);

        p ^= 1;
    }

#pragma unroll
    for (int r = 0; r < 2; ++r)
#pragma unroll
        for (int c = 0; c < 4; ++c)
#pragma unroll
            for (int i = 0; i < 8; ++i) {
                int m = m0 + wr * 32 + r * 16 + i + hi * 8;
                int n = n0 + wc * 64 + c * 16 + ln;
                out[(size_t)m * D_MODEL + n] = acc[r][c][i] + bias[n];
            }
}

// ===========================================================================
extern "C" void kernel_launch(void* const* d_in, const int* in_sizes, int n_in,
                              void* d_out, int out_size, void* d_ws, size_t ws_size,
                              hipStream_t stream) {
    const float* x    = (const float*)d_in[0];
    const float* Wqkv = (const float*)d_in[1];
    const float* bqkv = (const float*)d_in[2];
    const float* Wout = (const float*)d_in[3];
    const float* bout = (const float*)d_in[4];
    float* out = (float*)d_out;

    __bf16* qkvws   = (__bf16*)d_ws;                       // 3*B*H*T*hd bf16 (48 MB)
    __bf16* attnout = qkvws + 3 * SEL_STRIDE;              // B*T*C bf16 (16 MB)

    // 1) QKV projection (+bias) -> head-major bf16 workspace
    gemm_qkv_kernel<<<dim3(QKV_N / 128, M_TOTAL / 128), 256, 0, stream>>>(
        x, Wqkv, bqkv, qkvws);

    // 2) RoPE on Q and K (in place)
    int rope_threads = BATCH * N_HEADS * SEQ * 32;         // 4,194,304
    rope_kernel<<<rope_threads / 256, 256, 0, stream>>>(qkvws);
    rope_kernel<<<rope_threads / 256, 256, 0, stream>>>(qkvws + SEL_STRIDE);

    // 3) Flash attention -> [B,T,C] bf16
    attn_kernel<<<dim3(BATCH * N_HEADS, SEQ / 128), 256, 0, stream>>>(
        qkvws, attnout);

    // 4) Output projection (+bias) -> f32 d_out
    gemm_out_kernel<<<dim3(D_MODEL / 128, M_TOTAL / 128), 256, 0, stream>>>(
        attnout, Wout, bout, out);
}
